// BiAttentionFlow_68513318306103
// MI455X (gfx1250) — compile-verified, matrix operand order checked by reading, not run on hardware
//
#include <hip/hip_runtime.h>
#include <hip/hip_bf16.h>

typedef __attribute__((ext_vector_type(16))) __bf16 v16bf;
typedef __attribute__((ext_vector_type(8)))  float  v8f;
typedef __attribute__((ext_vector_type(4))) unsigned int u32x4;
typedef __attribute__((ext_vector_type(8))) int         i32x8;
typedef __attribute__((ext_vector_type(4))) int         i32x4;

#define NEG_INF_F (-1e30f)

// ---------- helpers ----------
__device__ __forceinline__ unsigned int pack2_bf16(float a, float b) {
  unsigned int ua = __float_as_uint(a);
  unsigned int ub = __float_as_uint(b);
  ua = (ua + 0x7FFFu + ((ua >> 16) & 1u)) >> 16;
  ub = (ub + 0x7FFFu + ((ub >> 16) & 1u)) >> 16;
  return (ua & 0xFFFFu) | (ub << 16);
}
__device__ __forceinline__ unsigned short f2bf(float a) {
  unsigned int u = __float_as_uint(a);
  return (unsigned short)((u + 0x7FFFu + ((u >> 16) & 1u)) >> 16);
}

union BF16Frag { unsigned int u[8]; v16bf v; };

// Problem constants
constexpr int BB = 16, LC = 4096, LQ = 512, DD = 128;

// LDS layout (in dwords)
constexpr int QLDS_W   = 65;                 // 64 bf16-pair dwords per q row + 1 pad
constexpr int QLDS_SZ  = LQ * QLDS_W;        // 33280
constexpr int QT_W     = 257;                // 256 dwords (512 q-rows bf16) + 1 pad
constexpr int QT_SZ    = DD * QT_W;          // 32896
constexpr int PLDS_WV  = 272;                // 16 rows * 17-dword padded stride
constexpr int PLDS_SZ  = 8 * PLDS_WV;        // 2176
constexpr int CW_SZ    = 128;                // 8 waves * 16 rows
constexpr int QW_SZ    = 512;
constexpr int WL_SZ    = 384;
constexpr int LDS_DW   = QLDS_SZ + QT_SZ + PLDS_SZ + CW_SZ + QW_SZ + WL_SZ;
constexpr size_t LDS_BYTES = (size_t)LDS_DW * 4;   // 277504 B < 320 KB WGP LDS

// ctx tile staged by TDM overlays the qlds region:
// 128 rows * (128 + 1 pad) dwords = 16512 dwords (66048 B)  <= QLDS_SZ
constexpr int CTILE_W = 129;

// ---------- kernel 0: qw[b,m] = q[b,m,:] . w_q + bias (+ mask) ----------
__global__ void __launch_bounds__(256)
qw_kernel(const float* __restrict__ q, const unsigned char* __restrict__ q_mask,
          const float* __restrict__ W, const float* __restrict__ bias,
          float* __restrict__ qw_g) {
  int idx = blockIdx.x * 256 + threadIdx.x;   // BB*LQ = 8192
  if (idx >= BB * LQ) return;
  const float* qr = q + (size_t)idx * DD;
  float acc = bias[0];
#pragma unroll 8
  for (int d = 0; d < DD; ++d) acc += qr[d] * W[DD + d];
  if (!q_mask[idx]) acc += NEG_INF_F;
  qw_g[idx] = acc;
}

// ---------- kernel 1: flash-style S + softmax + u, writes out[0..384) & rowmax ----------
__global__ void __launch_bounds__(256)
bidaf_main(const float* __restrict__ ctx, const float* __restrict__ q,
           const unsigned char* __restrict__ ctx_mask,
           const float* __restrict__ W, const float* __restrict__ qw_g,
           float* __restrict__ out, float* __restrict__ smax_g) {
  extern __shared__ unsigned int lds[];
  unsigned int* qlds = lds;                         // [512][65]  q row-major bf16 pairs (d pairs)
  unsigned int* qtl  = qlds + QLDS_SZ;              // [128][257] q d-major bf16 pairs (row pairs)
  unsigned int* plds = qtl + QT_SZ;                 // [8][272]   per-wave P scratch
  float* cwlds = (float*)(plds + PLDS_SZ);          // [8][16]
  float* qwlds = cwlds + CW_SZ;                     // [512]
  float* wlds  = qwlds + QW_SZ;                     // [384]

  const int tid  = threadIdx.x;
  const int wave = tid >> 5;
  const int lane = tid & 31;
  const int n16  = lane & 15;     // N / column index inside a 16-wide tile
  const int hh   = lane >> 4;     // lane half
  const int b    = blockIdx.x >> 5;        // 32 row-tiles per batch
  const int tile = blockIdx.x & 31;
  const int row0 = tile * 128 + wave * 16; // this wave owns rows [row0, row0+16)

  const float* qb = q + (size_t)b * LQ * DD;

  // ---- Phase A: TDM-stage the 128x128 fp32 ctx tile into LDS (padded rows) ----
  if (wave == 0) {
    unsigned long long ga =
        (unsigned long long)(const void*)(ctx + ((size_t)b * LC + (size_t)tile * 128) * DD);
    unsigned lva = (unsigned)(size_t)(void*)lds;   // LDS byte offset (low 32 bits of generic addr)
    // D# group0: count=1; lds_addr; 57-bit global addr; type=2 ("image")
    u32x4 g0 = { 1u, lva,
                 (unsigned)(ga & 0xFFFFFFFFu),
                 (unsigned)(((ga >> 32) & 0x01FFFFFFu) | (2u << 30)) };
    // D# group1: data_size=4B (2<<16), pad_enable (1<<20), pad_interval=6 (512B -> 6<<22),
    //            pad_amount=1 dword (0<<25); tensor_dim0=128, tensor_dim1=128,
    //            tile_dim0=128, tile_dim1=128, tile_dim2=0, dim0_stride=128
    i32x8 g1 = { (int)0x01920000u,
                 (int)(128u << 16),     // tensor_dim0[15:0] in bits 63:48
                 (int)(128u << 16),     // tensor_dim1[15:0] in bits 95:80
                 (int)(128u << 16),     // tile_dim0 in bits 127:112
                 (int)128,              // tile_dim1 in bits 143:128
                 (int)128,              // tensor_dim0_stride low 32
                 0, 0 };
    i32x4 gz = { 0, 0, 0, 0 };
#if defined(__clang_major__) && (__clang_major__ >= 23)
    i32x8 gz8 = { 0, 0, 0, 0, 0, 0, 0, 0 };
    __builtin_amdgcn_tensor_load_to_lds(g0, g1, gz, gz, gz8, 0);
#else
    __builtin_amdgcn_tensor_load_to_lds(g0, g1, gz, gz, 0);
#endif
    __builtin_amdgcn_s_wait_tensorcnt(0);
  }
  // Meanwhile-safe staging of small disjoint regions
  for (int i = tid; i < WL_SZ; i += 256) wlds[i] = W[i];
  for (int i = tid; i < LQ; i += 256)    qwlds[i] = qw_g[b * LQ + i];
  __syncthreads();   // ctx tile resident in LDS

  // ---- Phase B: build resident A fragments (ctx ⊙ w_m, bf16) + cw from the LDS tile ----
  const int myrow = row0 + n16;
  const float* crowl = (const float*)lds + (size_t)(wave * 16 + n16) * CTILE_W;
  BF16Frag areg[4];
  float cw_part = 0.f;
#pragma unroll
  for (int kc = 0; kc < 4; ++kc) {
#pragma unroll
    for (int v = 0; v < 8; ++v) {
      int d0 = kc * 32 + ((v >= 4) ? 16 : 0) + (v & 3) * 2 + hh * 8;
      float f0 = crowl[d0], f1 = crowl[d0 + 1];
      cw_part += f0 * wlds[d0] + f1 * wlds[d0 + 1];
      areg[kc].u[v] = pack2_bf16(f0 * wlds[256 + d0], f1 * wlds[256 + d0 + 1]);
    }
  }
  cw_part += __shfl_xor(cw_part, 16, 32);
  float cwfull = cw_part + (ctx_mask[(size_t)b * LC + myrow] ? 0.f : NEG_INF_F);
  if (hh == 0) cwlds[wave * 16 + n16] = cwfull;
  __syncthreads();   // everyone done reading the ctx tile; region can be reused

  float cwreg[8];
#pragma unroll
  for (int i = 0; i < 8; ++i) cwreg[i] = cwlds[wave * 16 + hh * 8 + i];

  // ---- Phase C: stage q (dual bf16 layouts), overwriting the ctx tile region ----
  unsigned short* qt16 = (unsigned short*)qtl;
  for (int idx = tid; idx < LQ * (DD / 2); idx += 256) {
    int r = idx >> 6, dp = idx & 63;
    float2 f = reinterpret_cast<const float2*>(qb)[idx];
    unsigned int p = pack2_bf16(f.x, f.y);
    qlds[r * QLDS_W + dp] = p;
    qt16[(2 * dp) * (2 * QT_W) + r]     = (unsigned short)(p & 0xFFFFu);
    qt16[(2 * dp + 1) * (2 * QT_W) + r] = (unsigned short)(p >> 16);
  }
  __syncthreads();

  // ---- Phase D: flash loop (online softmax over 16 chunks of 32 q columns) ----
  v8f utile[8];
#pragma unroll
  for (int t = 0; t < 8; ++t)
#pragma unroll
    for (int e = 0; e < 8; ++e) utile[t][e] = 0.f;
  float mrow[8], lrow[8];
#pragma unroll
  for (int i = 0; i < 8; ++i) { mrow[i] = -3.0e38f; lrow[i] = 0.f; }

  unsigned int*   myp   = plds + wave * PLDS_WV;
  unsigned short* myp16 = (unsigned short*)myp;

  for (int c = 0; c < 16; ++c) {
    v8f S[2];
#pragma unroll
    for (int j = 0; j < 2; ++j) {
      v8f acc;
#pragma unroll
      for (int e = 0; e < 8; ++e) acc[e] = 0.f;
      const int ncol = c * 32 + j * 16;
#pragma unroll
      for (int kc = 0; kc < 4; ++kc) {
        BF16Frag bf;
#pragma unroll
        for (int v = 0; v < 8; ++v) {
          int k = kc * 32 + ((v >= 4) ? 16 : 0) + (v & 3) * 2 + hh * 8;
          bf.u[v] = qlds[(ncol + n16) * QLDS_W + (k >> 1)];
        }
        acc = __builtin_amdgcn_wmma_f32_16x16x32_bf16(
                  false, areg[kc].v, false, bf.v, (short)0, acc, false, false);
      }
      float qwn = qwlds[ncol + n16];
#pragma unroll
      for (int e = 0; e < 8; ++e) acc[e] += cwreg[e] + qwn;
      S[j] = acc;
    }

    float scr[8];
#pragma unroll
    for (int i = 0; i < 8; ++i) {
      float v = fmaxf(S[0][i], S[1][i]);
      v = fmaxf(v, __shfl_xor(v, 1, 32));
      v = fmaxf(v, __shfl_xor(v, 2, 32));
      v = fmaxf(v, __shfl_xor(v, 4, 32));
      v = fmaxf(v, __shfl_xor(v, 8, 32));
      float mn = fmaxf(mrow[i], v);
      float sc = __expf(mrow[i] - mn);
      float p0 = __expf(S[0][i] - mn);
      float p1 = __expf(S[1][i] - mn);
      S[0][i] = p0; S[1][i] = p1;
      float rs = p0 + p1;
      rs += __shfl_xor(rs, 1, 32);
      rs += __shfl_xor(rs, 2, 32);
      rs += __shfl_xor(rs, 4, 32);
      rs += __shfl_xor(rs, 8, 32);
      lrow[i] = lrow[i] * sc + rs;
      mrow[i] = mn;
      scr[i] = sc;
    }
#pragma unroll
    for (int t = 0; t < 8; ++t)
#pragma unroll
      for (int i = 0; i < 8; ++i) utile[t][i] *= scr[i];

    // P (C-layout) -> per-wave LDS -> A-layout fragment (16x32)
#pragma unroll
    for (int i = 0; i < 8; ++i) {
      int r = i + 8 * hh;
      myp16[r * 34 + n16]      = f2bf(S[0][i]);
      myp16[r * 34 + 16 + n16] = f2bf(S[1][i]);
    }
    BF16Frag pf;
#pragma unroll
    for (int v = 0; v < 8; ++v) {
      int k = ((v >= 4) ? 16 : 0) + (v & 3) * 2 + hh * 8;
      pf.u[v] = myp[n16 * 17 + (k >> 1)];
    }
    // u += P @ q_chunk  (B operand from d-major qT layout)
#pragma unroll
    for (int t = 0; t < 8; ++t) {
      BF16Frag bq;
#pragma unroll
      for (int v = 0; v < 8; ++v) {
        int k = ((v >= 4) ? 16 : 0) + (v & 3) * 2 + hh * 8;   // q-row within chunk (even)
        bq.u[v] = qtl[(t * 16 + n16) * QT_W + ((c * 32 + k) >> 1)];
      }
      utile[t] = __builtin_amdgcn_wmma_f32_16x16x32_bf16(
                     false, pf.v, false, bq.v, (short)0, utile[t], false, false);
    }
  }

  // ---- Epilogue: rowmax to ws; out[:, :, 0:384) = [ctx, u, ctx*u] ----
  if (n16 == 0) {
#pragma unroll
    for (int i = 0; i < 8; ++i)
      smax_g[(size_t)b * LC + row0 + 8 * hh + i] = mrow[i];
  }
  float inv[8];
#pragma unroll
  for (int i = 0; i < 8; ++i) inv[i] = 1.f / lrow[i];
#pragma unroll
  for (int t = 0; t < 8; ++t) {
#pragma unroll
    for (int i = 0; i < 8; ++i) {
      int r = row0 + 8 * hh + i;
      int d = t * 16 + n16;
      float uv = utile[t][i] * inv[i];
      float cv = ctx[((size_t)b * LC + r) * DD + d];
      size_t ob = ((size_t)b * LC + r) * (4 * DD);
      out[ob + d]          = cv;
      out[ob + DD + d]     = uv;
      out[ob + 2 * DD + d] = cv * uv;
    }
  }
}

// ---------- kernel 2: per-batch q2c softmax over l, h = sum_l w_l * ctx[l,:] ----------
__global__ void __launch_bounds__(256)
hred_kernel(const float* __restrict__ ctx, const float* __restrict__ smax_g,
            float* __restrict__ h_g) {
  __shared__ float red[256];
  const int b = blockIdx.x, tid = threadIdx.x;
  const float* sm = smax_g + (size_t)b * LC;
  float mx = -3.0e38f;
  for (int l = tid; l < LC; l += 256) mx = fmaxf(mx, sm[l]);
  red[tid] = mx; __syncthreads();
  for (int s = 128; s > 0; s >>= 1) { if (tid < s) red[tid] = fmaxf(red[tid], red[tid + s]); __syncthreads(); }
  mx = red[0]; __syncthreads();
  float se = 0.f;
  for (int l = tid; l < LC; l += 256) se += __expf(sm[l] - mx);
  red[tid] = se; __syncthreads();
  for (int s = 128; s > 0; s >>= 1) { if (tid < s) red[tid] += red[tid + s]; __syncthreads(); }
  se = red[0]; __syncthreads();

  const int d = tid & 127, part = tid >> 7;
  const float* cb = ctx + (size_t)b * LC * DD;
  float acc = 0.f;
  for (int l = part; l < LC; l += 2) acc += __expf(sm[l] - mx) * cb[(size_t)l * DD + d];
  red[tid] = acc; __syncthreads();
  if (part == 0) h_g[b * DD + d] = (red[d] + red[d + 128]) / se;
}

// ---------- kernel 3: out[:, :, 384:512) = ctx * h (broadcast) ----------
__global__ void __launch_bounds__(256)
hmul_kernel(const float* __restrict__ ctx, const float* __restrict__ h_g,
            float* __restrict__ out) {
  size_t idx = (size_t)blockIdx.x * 256 + threadIdx.x;   // BB*LC*DD
  size_t bl = idx >> 7;
  int d = (int)(idx & 127);
  int b = (int)(bl >> 12);
  out[bl * (4 * DD) + 3 * DD + d] = ctx[idx] * h_g[b * DD + d];
}

// ---------- launch ----------
extern "C" void kernel_launch(void* const* d_in, const int* in_sizes, int n_in,
                              void* d_out, int out_size, void* d_ws, size_t ws_size,
                              hipStream_t stream) {
  const float* ctx  = (const float*)d_in[0];
  const float* q    = (const float*)d_in[1];
  const unsigned char* ctx_mask = (const unsigned char*)d_in[2];
  const unsigned char* q_mask   = (const unsigned char*)d_in[3];
  const float* W    = (const float*)d_in[4];
  const float* bias = (const float*)d_in[5];
  float* out = (float*)d_out;

  float* ws     = (float*)d_ws;
  float* qw_g   = ws;                    // BB*LQ   = 8192
  float* smax_g = ws + BB * LQ;          // BB*LC   = 65536
  float* h_g    = smax_g + BB * LC;      // BB*DD   = 2048

  qw_kernel<<<(BB * LQ + 255) / 256, 256, 0, stream>>>(q, q_mask, W, bias, qw_g);
  bidaf_main<<<BB * (LC / 128), 256, LDS_BYTES, stream>>>(ctx, q, ctx_mask, W, qw_g, out, smax_g);
  hred_kernel<<<BB, 256, 0, stream>>>(ctx, smax_g, h_g);
  hmul_kernel<<<(BB * LC * DD) / 256, 256, 0, stream>>>(ctx, h_g, out);
}
